// EncoderLayer_76106820485311
// MI455X (gfx1250) — matrix operand order repaired and checked by reading, not compiled, not run on hardware
//
#include <hip/hip_runtime.h>

// ---------------------------------------------------------------------------
// MI455X / gfx1250: bf16 WMMA GEMMs (v_wmma_f32_16x16x32_bf16), TDM async
// weight staging (tensor_load_to_lds + s_wait_tensorcnt), flash attention
// with online softmax, GShard top-2 MoE routing.
// ---------------------------------------------------------------------------

typedef __bf16 bf16;
typedef __attribute__((ext_vector_type(16))) __bf16 bf16x16;
typedef __attribute__((ext_vector_type(8)))  __bf16 bf16x8;
typedef __attribute__((ext_vector_type(8)))  float  floatx8;

#define B_SZ 8
#define N_SEQ 1024
#define D_MODEL 512
#define D_INNER 2048
#define N_HEAD 8
#define D_HEAD 64
#define N_EXP 24
#define N_GRP 4
#define T_GROUP 2048            // tokens per group = B * (N/4)
#define CAP 171                 // ceil(2048*2/24)
#define CAP_PAD 192             // padded to multiple of 64 for GEMM tiles

// Tensor Data Mover available? (5-arg builtin on ROCm7.2/clang-22,
// 6-arg on amdgpu-toolchain/clang-23; see CDNA5_HIP.md)
#if defined(__HIP_DEVICE_COMPILE__) && defined(__gfx1250__) && \
    __has_builtin(__builtin_amdgcn_tensor_load_to_lds) &&      \
    __has_builtin(__builtin_amdgcn_s_wait_tensorcnt)
#define USE_TDM 1
#else
#define USE_TDM 0
#endif

#if USE_TDM
typedef __attribute__((ext_vector_type(4))) unsigned int u32x4;
typedef __attribute__((ext_vector_type(8))) int i32x8;
typedef __attribute__((ext_vector_type(4))) int i32x4;

// Issue a TDM load of a 2-D bf16 tile [tile_k rows x tile_n elems] from a
// row-major tensor (row length n_len elems, k_len rows) into LDS.
// D# packing per CDNA5 ISA.
__device__ inline void tdm_load_tile_bf16(unsigned lds_off, const bf16* gptr,
                                          int n_len, int k_len,
                                          int tile_n, int tile_k)
{
  const unsigned long long ga = (unsigned long long)(size_t)gptr;
  u32x4 g0;
  g0[0] = 1u;                                          // count=1, user D#
  g0[1] = lds_off;                                     // lds_addr
  g0[2] = (unsigned)(ga & 0xFFFFFFFFull);              // global_addr[31:0]
  g0[3] = (unsigned)((ga >> 32) & 0x01FFFFFFull) | (2u << 30);  // addr hi | type=2
  i32x8 g1;
  g1[0] = (1 << 16);                                   // data_size=1 (2 bytes)
  g1[1] = (int)(((unsigned)n_len & 0xFFFFu) << 16);    // tensor_dim0[15:0]
  g1[2] = (int)(((unsigned)n_len >> 16) |
                (((unsigned)k_len & 0xFFFFu) << 16));  // dim0 hi | tensor_dim1 lo
  g1[3] = (int)(((unsigned)k_len >> 16) |
                ((unsigned)tile_n << 16));             // dim1 hi | tile_dim0
  g1[4] = tile_k & 0xFFFF;                             // tile_dim1 (tile_dim2=0)
  g1[5] = n_len;                                       // tensor_dim0_stride[31:0]
  g1[6] = 0;                                           // stride hi | dim1_stride lo
  g1[7] = 0;
  i32x4 gz = {0, 0, 0, 0};
#if __clang_major__ >= 23
  i32x8 gz8 = {0, 0, 0, 0, 0, 0, 0, 0};
  __builtin_amdgcn_tensor_load_to_lds(g0, g1, gz, gz, gz8, 0);
#else
  __builtin_amdgcn_tensor_load_to_lds(g0, g1, gz, gz, 0);
#endif
}
#endif // USE_TDM

// ---------------------------------------------------------------------------
// WMMA helpers (CDNA5 wave32 layouts, CDNA5 ISA §7.12.2)
// ---------------------------------------------------------------------------
__device__ inline floatx8 wmma_bf16(bf16x16 a, bf16x16 b, floatx8 c) {
  return __builtin_amdgcn_wmma_f32_16x16x32_bf16(false, a, false, b, (short)0, c,
                                                 false, false);
}

// A-fragment 16x32 bf16 from row-major LDS: per lane two contiguous 16B runs.
__device__ inline bf16x16 load_a_frag(const bf16* smem, int ldm, int row_base, int k_base) {
  const int lane = threadIdx.x & 31;
  const int half = (lane >> 4) & 1;
  const bf16* p = smem + (long)(row_base + (lane & 15)) * ldm + k_base + half * 16;
  const bf16x8 lo = *(const bf16x8*)(p);        // K = half*16 + 0..7
  const bf16x8 hi = *(const bf16x8*)(p + 8);    // K = half*16 + 8..15
  return __builtin_shufflevector(lo, hi, 0, 1, 2, 3, 4, 5, 6, 7,
                                 8, 9, 10, 11, 12, 13, 14, 15);
}

// B-fragment from packed fragment-order LDS region [32 lanes][16 elems]:
// one contiguous 32B run per lane (2x ds_load_b128, no packing movs).
__device__ inline bf16x16 load_b_packed(const bf16* region) {
  return *(const bf16x16*)(region + (threadIdx.x & 31) * 16);
}

// scatter one B element (k within 0..31 of the k-step, n within 0..15 of the
// 16-wide tile) to fragment order: lane = (n&15) + 16*((k>>4)&1), elem = k&15
#define BFRAG_LANE(k, n) (((n) & 15) + 16 * (((k) >> 4) & 1))
#define BFRAG_ELEM(k)    ((k) & 15)

// ---------------------------------------------------------------------------
// fp32 -> bf16 conversion (weights)
// ---------------------------------------------------------------------------
__global__ void cvt_bf16_kernel(const float* __restrict__ src, bf16* __restrict__ dst, long n) {
  long i = (long)blockIdx.x * blockDim.x + threadIdx.x;
  long stride = (long)gridDim.x * blockDim.x;
  for (; i < n; i += stride) dst[i] = (bf16)src[i];
}

// ---------------------------------------------------------------------------
// Generic tiled GEMM: C[M,N](f32) = act(A[M,K](f32) @ Bw[K,N](bf16) + bias) (+resid)
// 64x64 tile, 128 threads = 4 waves in 2x2, each wave 2x2 WMMA tiles.
// B tile is double-buffered through the Tensor Data Mover (raw [32][64]),
// then swizzled into WMMA fragment order in LDS.
// Batched over blockIdx.z (expert slabs): B/bias indexed modulo bMod.
// ---------------------------------------------------------------------------
__global__ __launch_bounds__(128)
void gemm_bf16_kernel(const float* __restrict__ A, const bf16* __restrict__ Bw,
                      const float* __restrict__ bias, const float* __restrict__ resid,
                      float* __restrict__ C,
                      int N, int K,
                      long strideA, long strideB, int bMod, long strideBias, long strideC,
                      int act)
{
  const int zb = blockIdx.z;
  A += (long)zb * strideA;
  C += (long)zb * strideC;
  const int ze = bMod > 1 ? (zb % bMod) : 0;
  Bw += (long)ze * strideB;
  if (bias) bias += (long)ze * strideBias;

  __shared__ __align__(16) bf16 sA[64][40];       // 64 rows x 32 k (+pad, 80B rows)
  __shared__ __align__(16) bf16 sBf[4][32][16];   // fragment-order B (4 n-tiles)
#if USE_TDM
  __shared__ __align__(16) bf16 sBraw[2][32][64]; // TDM double buffer (raw [k][n])
#endif

  const int tid = threadIdx.x;
  const int wid = tid >> 5;
  const int lane = tid & 31;
  const int wr = wid >> 1, wc = wid & 1;
  const int row0 = blockIdx.y * 64;
  const int col0 = blockIdx.x * 64;

  floatx8 acc[2][2] = {};

#if USE_TDM
  if (wid == 0)
    tdm_load_tile_bf16((unsigned)(size_t)&sBraw[0][0][0], Bw + col0, N, K, 64, 32);
#endif

  for (int k0 = 0; k0 < K; k0 += 32) {
    const int cur = (k0 >> 5) & 1;
    __syncthreads();   // previous compute done: sA/sBf free, raw[cur^1] reads done
#if USE_TDM
    if (k0 + 32 < K && wid == 0)
      tdm_load_tile_bf16((unsigned)(size_t)&sBraw[cur ^ 1][0][0],
                         Bw + (long)(k0 + 32) * N + col0, N, K, 64, 32);
#endif
    // stage A tile (f32 -> bf16): 2048 elems as 512 float4
    for (int i = tid; i < 64 * 8; i += 128) {
      int r = i >> 3, c4 = (i & 7) * 4;
      const float4 v = *(const float4*)(A + (long)(row0 + r) * K + k0 + c4);
      sA[r][c4 + 0] = (bf16)v.x; sA[r][c4 + 1] = (bf16)v.y;
      sA[r][c4 + 2] = (bf16)v.z; sA[r][c4 + 3] = (bf16)v.w;
    }
    if (k0 + 32 < K)
      __builtin_prefetch(A + (long)(row0 + tid % 64) * K + k0 + 32, 0, 1);
#if USE_TDM
    if (wid == 0) {
      if (k0 + 32 < K) __builtin_amdgcn_s_wait_tensorcnt((short)1);  // cur done, nxt in flight
      else             __builtin_amdgcn_s_wait_tensorcnt((short)0);
    }
#endif
    __syncthreads();   // raw[cur] visible to all waves, sA staged

    // swizzle B tile into WMMA fragment order
    for (int i = tid; i < 256; i += 128) {
      const int k = i >> 3, n0 = (i & 7) * 8;
#if USE_TDM
      uint4 chunk = *(const uint4*)&sBraw[cur][k][n0];
#else
      uint4 chunk = *(const uint4*)(Bw + (long)(k0 + k) * N + col0 + n0);
#endif
      const bf16* e8 = (const bf16*)&chunk;
      const int elem = BFRAG_ELEM(k);
      const int lh = 16 * ((k >> 4) & 1);
#pragma unroll
      for (int j = 0; j < 8; ++j) {
        const int n = n0 + j;
        sBf[n >> 4][(n & 15) + lh][elem] = e8[j];
      }
    }
    __syncthreads();

    bf16x16 a0 = load_a_frag(&sA[0][0], 40, wr * 32 + 0,  0);
    bf16x16 a1 = load_a_frag(&sA[0][0], 40, wr * 32 + 16, 0);
    bf16x16 b0 = load_b_packed(&sBf[wc * 2 + 0][0][0]);
    bf16x16 b1 = load_b_packed(&sBf[wc * 2 + 1][0][0]);
    acc[0][0] = wmma_bf16(a0, b0, acc[0][0]);
    acc[0][1] = wmma_bf16(a0, b1, acc[0][1]);
    acc[1][0] = wmma_bf16(a1, b0, acc[1][0]);
    acc[1][1] = wmma_bf16(a1, b1, acc[1][1]);
  }

  const int half = lane >> 4, ln = lane & 15;
#pragma unroll
  for (int mi = 0; mi < 2; ++mi)
#pragma unroll
    for (int ni = 0; ni < 2; ++ni) {
      const int col = col0 + wc * 32 + ni * 16 + ln;
      const float bv = bias ? bias[col] : 0.f;
#pragma unroll
      for (int r = 0; r < 8; ++r) {
        const int row = row0 + wr * 32 + mi * 16 + half * 8 + r;
        float v = acc[mi][ni][r] + bv;
        if (act) v = fmaxf(v, 0.f);                  // relu(x@wo) ...
        if (resid) v += resid[(long)row * N + col];  // ... + residual
        C[(long)row * N + col] = v;
      }
    }
}

// ---------------------------------------------------------------------------
// Flash attention: grid(N/64, B*NH), 128 threads = 4 waves; each wave owns 16
// query rows. S = (Q*scale) @ K^T via WMMA, online softmax, O += P @ V via WMMA.
// K^T and V tiles are staged in WMMA B-fragment order (contiguous lane reads).
// ---------------------------------------------------------------------------
__global__ __launch_bounds__(128)
void attn_kernel(const float* __restrict__ Q, const float* __restrict__ Kb,
                 const float* __restrict__ Vb, float* __restrict__ O)
{
  const int bh = blockIdx.y;
  const int b_ = bh / N_HEAD, h = bh % N_HEAD;
  const int q0 = blockIdx.x * 64;
  const int tid = threadIdx.x, wid = tid >> 5, lane = tid & 31;
  const int half = lane >> 4, ln = lane & 15;

  __shared__ __align__(16) bf16 sQ[64][72];         // A-operand (row-major, 144B rows)
  __shared__ __align__(16) bf16 sKf[2][4][32][16];  // K^T in B-fragment order
  __shared__ __align__(16) bf16 sVf[2][4][32][16];  // V   in B-fragment order
  __shared__ __align__(16) bf16 sP[4][16][72];      // per-wave P tile (A-operand)

  const long rowbase = (long)b_ * N_SEQ;
  const int coff = h * D_HEAD;
  const float scale = 0.125f;                       // 1/sqrt(64)

  for (int i = tid; i < 64 * 64; i += 128) {
    int r = i >> 6, c = i & 63;
    sQ[r][c] = (bf16)(Q[(rowbase + q0 + r) * D_MODEL + coff + c] * scale);
  }

  floatx8 o[4] = {};
  float mrow[8], srow[8];
#pragma unroll
  for (int r = 0; r < 8; ++r) { mrow[r] = -1e30f; srow[r] = 0.f; }

  for (int j0 = 0; j0 < N_SEQ; j0 += 64) {
    for (int i = tid; i < 64 * 64; i += 128) {
      const int r = i >> 6, c = i & 63;   // r = key row in tile, c = head dim
      const float kv = Kb[(rowbase + j0 + r) * D_MODEL + coff + c];
      const float vv = Vb[(rowbase + j0 + r) * D_MODEL + coff + c];
      // S = Q @ K^T : B-operand (n = key r, k = dim c)
      sKf[c >> 5][r >> 4][BFRAG_LANE(c, r)][BFRAG_ELEM(c)] = (bf16)kv;
      // O = P @ V   : B-operand (n = dim c, k = key r)
      sVf[r >> 5][c >> 4][BFRAG_LANE(r, c)][BFRAG_ELEM(r)] = (bf16)vv;
    }
    __syncthreads();

    floatx8 s[4] = {};
#pragma unroll
    for (int ks = 0; ks < 2; ++ks) {
      bf16x16 aq = load_a_frag(&sQ[0][0], 72, wid * 16, ks * 32);
#pragma unroll
      for (int ni = 0; ni < 4; ++ni) {
        bf16x16 bk = load_b_packed(&sKf[ks][ni][0][0]);
        s[ni] = wmma_bf16(aq, bk, s[ni]);
      }
    }

    // online softmax; this lane's stats row = half*8 + r
#pragma unroll
    for (int r = 0; r < 8; ++r) {
      float lm = s[0][r];
#pragma unroll
      for (int ni = 1; ni < 4; ++ni) lm = fmaxf(lm, s[ni][r]);
#pragma unroll
      for (int off = 1; off < 16; off <<= 1)
        lm = fmaxf(lm, __shfl_xor(lm, off, 32));   // reduce within 16-lane half
      const float mnew  = fmaxf(mrow[r], lm);
      const float alpha = __expf(mrow[r] - mnew);
      float psum = 0.f;
#pragma unroll
      for (int ni = 0; ni < 4; ++ni) {
        float p = __expf(s[ni][r] - mnew);
        s[ni][r] = p;
        psum += p;
      }
#pragma unroll
      for (int off = 1; off < 16; off <<= 1)
        psum += __shfl_xor(psum, off, 32);
      srow[r] = srow[r] * alpha + psum;
      mrow[r] = mnew;
#pragma unroll
      for (int ni = 0; ni < 4; ++ni) o[ni][r] *= alpha;
    }

    // stage P (A-operand of P@V) through per-wave LDS
#pragma unroll
    for (int ni = 0; ni < 4; ++ni)
#pragma unroll
      for (int r = 0; r < 8; ++r)
        sP[wid][half * 8 + r][ni * 16 + ln] = (bf16)s[ni][r];
    __syncthreads();

#pragma unroll
    for (int ks = 0; ks < 2; ++ks) {
      bf16x16 ap = load_a_frag(&sP[wid][0][0], 72, 0, ks * 32);
#pragma unroll
      for (int ni = 0; ni < 4; ++ni) {
        bf16x16 bv = load_b_packed(&sVf[ks][ni][0][0]);
        o[ni] = wmma_bf16(ap, bv, o[ni]);
      }
    }
    __syncthreads();
  }

#pragma unroll
  for (int ni = 0; ni < 4; ++ni)
#pragma unroll
    for (int r = 0; r < 8; ++r) {
      const int qrow = q0 + wid * 16 + half * 8 + r;
      const int c = ni * 16 + ln;
      O[(rowbase + qrow) * D_MODEL + coff + c] = o[ni][r] / srow[r];
    }
}

// ---------------------------------------------------------------------------
// LayerNorm over last dim. gamma/beta selected per row group (seldiv) from up
// to 4 pointer sets (ln1..ln4); act=1 applies relu after affine.
// ---------------------------------------------------------------------------
__global__ __launch_bounds__(256)
void ln_kernel(const float* __restrict__ in, float* __restrict__ out,
               const float* g0, const float* g1, const float* g2, const float* g3,
               const float* b0, const float* b1, const float* b2, const float* b3,
               int D, int seldiv, int act)
{
  const long row = blockIdx.x;
  const int sel = (int)(row / seldiv) & 3;
  const float* gp = sel == 0 ? g0 : sel == 1 ? g1 : sel == 2 ? g2 : g3;
  const float* bp = sel == 0 ? b0 : sel == 1 ? b1 : sel == 2 ? b2 : b3;
  const float* p = in + row * (long)D;
  float* q = out + row * (long)D;

  __shared__ float red[256];
  float s = 0.f, ss = 0.f;
  for (int c = threadIdx.x; c < D; c += 256) { float v = p[c]; s += v; ss += v * v; }
  red[threadIdx.x] = s; __syncthreads();
  for (int o = 128; o > 0; o >>= 1) { if (threadIdx.x < o) red[threadIdx.x] += red[threadIdx.x + o]; __syncthreads(); }
  const float mean = red[0] / D; __syncthreads();
  red[threadIdx.x] = ss; __syncthreads();
  for (int o = 128; o > 0; o >>= 1) { if (threadIdx.x < o) red[threadIdx.x] += red[threadIdx.x + o]; __syncthreads(); }
  const float var = red[0] / D - mean * mean;
  const float rstd = rsqrtf(var + 1e-6f);
  for (int c = threadIdx.x; c < D; c += 256) {
    float v = (p[c] - mean) * rstd * gp[c] + bp[c];
    if (act) v = fmaxf(v, 0.f);
    q[c] = v;
  }
}

// ---------------------------------------------------------------------------
// ctx[B,N,D] -> grouped xf[G][2048][D]  (t = b*256 + r ; n = g*256 + r)
// ---------------------------------------------------------------------------
__global__ void group_reorder_kernel(const float* __restrict__ src, float* __restrict__ dst)
{
  long i = (long)blockIdx.x * blockDim.x + threadIdx.x;
  const long total = (long)N_GRP * T_GROUP * D_MODEL;
  if (i >= total) return;
  const int d = (int)(i & (D_MODEL - 1));
  const long row = i >> 9;                 // g*2048 + t
  const int g = (int)(row >> 11);
  const int t = (int)(row & 2047);
  const int b_ = t >> 8, r = t & 255;
  const long srow = (long)b_ * N_SEQ + g * 256 + r;
  dst[i] = src[srow * D_MODEL + d];
}

// ---------------------------------------------------------------------------
// GShard top-2 routing, one block per group. Exact reference slot semantics:
// second-choice slots start at full first-choice counts.
// ---------------------------------------------------------------------------
__global__ __launch_bounds__(256)
void routing_kernel(const float* __restrict__ xf, const float* __restrict__ wg,
                    int* __restrict__ e1, int* __restrict__ e2,
                    float* __restrict__ wn1, float* __restrict__ wn2,
                    int* __restrict__ slot1, int* __restrict__ slot2,
                    int* __restrict__ tokmap)
{
  const int g = blockIdx.x;
  __shared__ int sE1[T_GROUP];
  __shared__ int sE2[T_GROUP];
  const float* xg = xf + (long)g * T_GROUP * D_MODEL;

  for (int t = threadIdx.x; t < T_GROUP; t += 256) {
    float logits[N_EXP];
#pragma unroll
    for (int e = 0; e < N_EXP; ++e) logits[e] = 0.f;
    const float* xp = xg + (long)t * D_MODEL;
    for (int d = 0; d < D_MODEL; ++d) {
      const float xv = xp[d];
      const float* wrow = wg + d * N_EXP;
#pragma unroll
      for (int e = 0; e < N_EXP; ++e) logits[e] += xv * wrow[e];
    }
    float mx = logits[0];
#pragma unroll
    for (int e = 1; e < N_EXP; ++e) mx = fmaxf(mx, logits[e]);
    float sum = 0.f;
#pragma unroll
    for (int e = 0; e < N_EXP; ++e) { logits[e] = __expf(logits[e] - mx); sum += logits[e]; }
    int a = 0; float v1 = -1.f;
#pragma unroll
    for (int e = 0; e < N_EXP; ++e) if (logits[e] > v1) { v1 = logits[e]; a = e; }
    int bch = -1; float v2 = -1.f;
#pragma unroll
    for (int e = 0; e < N_EXP; ++e) if (e != a && logits[e] > v2) { v2 = logits[e]; bch = e; }
    const float p1 = v1 / sum, p2 = v2 / sum, ws = p1 + p2;
    sE1[t] = a; sE2[t] = bch;
    const long gt = (long)g * T_GROUP + t;
    e1[gt] = a; e2[gt] = bch; wn1[gt] = p1 / ws; wn2[gt] = p2 / ws;
  }
  for (int i = threadIdx.x; i < N_EXP * CAP_PAD; i += 256)
    tokmap[(long)g * N_EXP * CAP_PAD + i] = -1;
  __syncthreads();

  if (threadIdx.x == 0) {
    int c[N_EXP];
#pragma unroll
    for (int e = 0; e < N_EXP; ++e) c[e] = 0;
    for (int t = 0; t < T_GROUP; ++t) {           // first-choice pass
      const int a = sE1[t];
      const int s = c[a]; c[a] = s + 1;
      const int sv = (s < CAP) ? s : -1;
      slot1[(long)g * T_GROUP + t] = sv;
      if (sv >= 0) tokmap[((long)g * N_EXP + a) * CAP_PAD + sv] = t;
    }
    for (int t = 0; t < T_GROUP; ++t) {           // second-choice pass
      const int a = sE2[t];
      const int s = c[a]; c[a] = s + 1;
      const int sv = (s < CAP) ? s : -1;
      slot2[(long)g * T_GROUP + t] = sv;
      if (sv >= 0) tokmap[((long)g * N_EXP + a) * CAP_PAD + sv] = t;
    }
  }
}

// ---------------------------------------------------------------------------
// Gather token rows into expert-slot slabs (zero for empty slots)
// ---------------------------------------------------------------------------
__global__ __launch_bounds__(256)
void gather_kernel(const float* __restrict__ src, const int* __restrict__ tokmap,
                   float* __restrict__ dst, int D)
{
  const long row = blockIdx.x;                    // G*24*CAP_PAD rows
  const int g = (int)(row / (N_EXP * CAP_PAD));
  const int tok = tokmap[row];
  float* dp = dst + row * (long)D;
  if (tok < 0) {
    for (int c = threadIdx.x; c < D; c += 256) dp[c] = 0.f;
  } else {
    const float* sp = src + ((long)g * T_GROUP + tok) * D;
    for (int c = threadIdx.x; c < D; c += 256) dp[c] = sp[c];
  }
}

// ---------------------------------------------------------------------------
// Weighted combine of two expert-slot rows back to token space
// ---------------------------------------------------------------------------
__global__ __launch_bounds__(256)
void combine_kernel(const float* __restrict__ eo,
                    const int* __restrict__ e1, const int* __restrict__ e2,
                    const int* __restrict__ slot1, const int* __restrict__ slot2,
                    const float* __restrict__ wn1, const float* __restrict__ wn2,
                    float* __restrict__ dst, int D)
{
  const long gt = blockIdx.x;                     // G*2048 token rows
  const int g = (int)(gt / T_GROUP);
  const int a = e1[gt], b = e2[gt];
  const int s1 = slot1[gt], s2 = slot2[gt];
  const float w1 = (s1 >= 0) ? wn1[gt] : 0.f;
  const float w2 = (s2 >= 0) ? wn2[gt] : 0.f;
  const float* p1 = eo + (((long)g * N_EXP + a) * CAP_PAD + (s1 < 0 ? 0 : s1)) * D;
  const float* p2 = eo + (((long)g * N_EXP + b) * CAP_PAD + (s2 < 0 ? 0 : s2)) * D;
  float* dp = dst + gt * (long)D;
  for (int c = threadIdx.x; c < D; c += 256) dp[c] = w1 * p1[c] + w2 * p2[c];
}

// ---------------------------------------------------------------------------
// Ungroup + residual + final LayerNorm -> d_out
// ---------------------------------------------------------------------------
__global__ __launch_bounds__(256)
void final_ln_kernel(const float* __restrict__ yg, const float* __restrict__ resid,
                     const float* __restrict__ gam, const float* __restrict__ bet,
                     float* __restrict__ out)
{
  const int row = blockIdx.x;                     // 0..8191 (b*1024+n)
  const int b_ = row >> 10, n = row & 1023;
  const int g = n >> 8, r = n & 255;
  const long t = (long)g * T_GROUP + b_ * 256 + r;
  const float* py = yg + t * D_MODEL;
  const float* pr = resid + (long)row * D_MODEL;

  __shared__ float red[256];
  __shared__ float buf[D_MODEL];
  float s = 0.f, ss = 0.f;
  for (int c = threadIdx.x; c < D_MODEL; c += 256) {
    float v = py[c] + pr[c];
    buf[c] = v; s += v; ss += v * v;
  }
  red[threadIdx.x] = s; __syncthreads();
  for (int o = 128; o > 0; o >>= 1) { if (threadIdx.x < o) red[threadIdx.x] += red[threadIdx.x + o]; __syncthreads(); }
  const float mean = red[0] / D_MODEL; __syncthreads();
  red[threadIdx.x] = ss; __syncthreads();
  for (int o = 128; o > 0; o >>= 1) { if (threadIdx.x < o) red[threadIdx.x] += red[threadIdx.x + o]; __syncthreads(); }
  const float var = red[0] / D_MODEL - mean * mean;
  const float rstd = rsqrtf(var + 1e-6f);
  for (int c = threadIdx.x; c < D_MODEL; c += 256)
    out[(long)row * D_MODEL + c] = (buf[c] - mean) * rstd * gam[c] + bet[c];
}

// ---------------------------------------------------------------------------
// Host orchestration
// ---------------------------------------------------------------------------
extern "C" void kernel_launch(void* const* d_in, const int* in_sizes, int n_in,
                              void* d_out, int out_size, void* d_ws, size_t ws_size,
                              hipStream_t stream)
{
  (void)in_sizes; (void)n_in; (void)out_size; (void)ws_size;
  const float* x        = (const float*)d_in[0];
  const float* wq       = (const float*)d_in[1];
  const float* wk       = (const float*)d_in[2];
  const float* wv       = (const float*)d_in[3];
  const float* wo       = (const float*)d_in[4];
  const float* ln_att_g = (const float*)d_in[5];
  const float* ln_att_b = (const float*)d_in[6];
  const float* wg1      = (const float*)d_in[7];
  const float* w1       = (const float*)d_in[8];
  const float* b1       = (const float*)d_in[9];
  /* d_in[10] = wg2 (unused by reference math) */
  const float* w2       = (const float*)d_in[11];
  const float* b2       = (const float*)d_in[12];
  const float* ln_out_g = (const float*)d_in[13];
  const float* ln_out_b = (const float*)d_in[14];
  const float* ln_g[4]  = { (const float*)d_in[15], (const float*)d_in[17],
                            (const float*)d_in[19], (const float*)d_in[21] };
  const float* ln_b[4]  = { (const float*)d_in[16], (const float*)d_in[18],
                            (const float*)d_in[20], (const float*)d_in[22] };
  float* out = (float*)d_out;

  const long ROWS   = (long)B_SZ * N_SEQ;                 // 8192
  const long NSLOT  = (long)N_GRP * N_EXP * CAP_PAD;      // 18432
  const long NTOK   = (long)N_GRP * T_GROUP;              // 8192

  char* ws = (char*)d_ws;
  size_t off = 0;
  auto alloc = [&](size_t bytes) -> void* {
    void* p = ws + off;
    off = (off + bytes + 255) & ~(size_t)255;
    return p;
  };
  bf16*  wq_bf  = (bf16*)alloc((size_t)D_MODEL * D_MODEL * 2);
  bf16*  wk_bf  = (bf16*)alloc((size_t)D_MODEL * D_MODEL * 2);
  bf16*  wv_bf  = (bf16*)alloc((size_t)D_MODEL * D_MODEL * 2);
  bf16*  wo_bf  = (bf16*)alloc((size_t)D_MODEL * D_MODEL * 2);
  bf16*  w1_bf  = (bf16*)alloc((size_t)N_EXP * D_MODEL * D_INNER * 2);
  bf16*  w2_bf  = (bf16*)alloc((size_t)N_EXP * D_INNER * D_MODEL * 2);
  float* qb     = (float*)alloc((size_t)ROWS * D_MODEL * 4);
  float* kb     = (float*)alloc((size_t)ROWS * D_MODEL * 4);
  float* vb     = (float*)alloc((size_t)ROWS * D_MODEL * 4);
  float* ctxb   = (float*)alloc((size_t)ROWS * D_MODEL * 4);   // attention output
  float* ctx_ln = (float*)alloc((size_t)ROWS * D_MODEL * 4);   // post-attn LN (resid)
  float* xf     = (float*)alloc((size_t)NTOK * D_MODEL * 4);   // grouped tokens
  float* y_g    = (float*)alloc((size_t)NTOK * D_MODEL * 4);   // grouped MoE output
  int*   e1     = (int*)alloc((size_t)NTOK * 4);
  int*   e2     = (int*)alloc((size_t)NTOK * 4);
  int*   slot1  = (int*)alloc((size_t)NTOK * 4);
  int*   slot2  = (int*)alloc((size_t)NTOK * 4);
  float* wn1    = (float*)alloc((size_t)NTOK * 4);
  float* wn2    = (float*)alloc((size_t)NTOK * 4);
  int*   tokmap = (int*)alloc((size_t)NSLOT * 4);
  float* xe     = (float*)alloc((size_t)NSLOT * D_MODEL * 4);  // expert in/out (D_MODEL)
  float* he     = (float*)alloc((size_t)NSLOT * D_INNER * 4);  // expert h slabs (reused)
  float* htok   = (float*)alloc((size_t)NTOK * D_INNER * 4);   // token-space h

  // 1) weights -> bf16
  cvt_bf16_kernel<<<1024, 256, 0, stream>>>(wq, wq_bf, (long)D_MODEL * D_MODEL);
  cvt_bf16_kernel<<<1024, 256, 0, stream>>>(wk, wk_bf, (long)D_MODEL * D_MODEL);
  cvt_bf16_kernel<<<1024, 256, 0, stream>>>(wv, wv_bf, (long)D_MODEL * D_MODEL);
  cvt_bf16_kernel<<<1024, 256, 0, stream>>>(wo, wo_bf, (long)D_MODEL * D_MODEL);
  cvt_bf16_kernel<<<4096, 256, 0, stream>>>(w1, w1_bf, (long)N_EXP * D_MODEL * D_INNER);
  cvt_bf16_kernel<<<4096, 256, 0, stream>>>(w2, w2_bf, (long)N_EXP * D_INNER * D_MODEL);

  // 2) QKV projections: [8192,512] @ [512,512]
  {
    dim3 grid(D_MODEL / 64, (unsigned)(ROWS / 64), 1);
    gemm_bf16_kernel<<<grid, 128, 0, stream>>>(x, wq_bf, nullptr, nullptr, qb,
        D_MODEL, D_MODEL, 0, 0, 1, 0, 0, 0);
    gemm_bf16_kernel<<<grid, 128, 0, stream>>>(x, wk_bf, nullptr, nullptr, kb,
        D_MODEL, D_MODEL, 0, 0, 1, 0, 0, 0);
    gemm_bf16_kernel<<<grid, 128, 0, stream>>>(x, wv_bf, nullptr, nullptr, vb,
        D_MODEL, D_MODEL, 0, 0, 1, 0, 0, 0);
  }

  // 3) flash attention
  attn_kernel<<<dim3(N_SEQ / 64, B_SZ * N_HEAD), 128, 0, stream>>>(qb, kb, vb, ctxb);

  // 4) output projection: relu(ctx@wo) + x  (written into qb, then LN -> ctx_ln)
  {
    dim3 grid(D_MODEL / 64, (unsigned)(ROWS / 64), 1);
    gemm_bf16_kernel<<<grid, 128, 0, stream>>>(ctxb, wo_bf, nullptr, x, qb,
        D_MODEL, D_MODEL, 0, 0, 1, 0, 0, /*relu=*/1);
  }
  ln_kernel<<<(unsigned)ROWS, 256, 0, stream>>>(qb, ctx_ln,
      ln_att_g, ln_att_g, ln_att_g, ln_att_g,
      ln_att_b, ln_att_b, ln_att_b, ln_att_b, D_MODEL, 1 << 30, 0);

  // 5) group reorder, routing
  {
    const long total = (long)N_GRP * T_GROUP * D_MODEL;
    group_reorder_kernel<<<(unsigned)((total + 255) / 256), 256, 0, stream>>>(ctx_ln, xf);
  }
  routing_kernel<<<N_GRP, 256, 0, stream>>>(xf, wg1, e1, e2, wn1, wn2, slot1, slot2, tokmap);

  // 6) MoE layer 1: gather -> [96 slabs of 192x512] @ w1[e] -> he
  gather_kernel<<<(unsigned)NSLOT, 256, 0, stream>>>(xf, tokmap, xe, D_MODEL);
  {
    dim3 grid(D_INNER / 64, CAP_PAD / 64, N_GRP * N_EXP);
    gemm_bf16_kernel<<<grid, 128, 0, stream>>>(xe, w1_bf, b1, nullptr, he,
        D_INNER, D_MODEL,
        (long)CAP_PAD * D_MODEL, (long)D_MODEL * D_INNER, N_EXP,
        (long)D_INNER, (long)CAP_PAD * D_INNER, 0);
  }
  combine_kernel<<<(unsigned)NTOK, 256, 0, stream>>>(he, e1, e2, slot1, slot2, wn1, wn2,
                                                     htok, D_INNER);

  // 7) per-group LN + relu on token-space h
  ln_kernel<<<(unsigned)NTOK, 256, 0, stream>>>(htok, htok,
      ln_g[0], ln_g[1], ln_g[2], ln_g[3],
      ln_b[0], ln_b[1], ln_b[2], ln_b[3], D_INNER, T_GROUP, /*relu=*/1);

  // 8) MoE layer 2: re-gather (reuse he) -> [192x2048] @ w2[e] -> xe (reuse)
  gather_kernel<<<(unsigned)NSLOT, 256, 0, stream>>>(htok, tokmap, he, D_INNER);
  {
    dim3 grid(D_MODEL / 64, CAP_PAD / 64, N_GRP * N_EXP);
    gemm_bf16_kernel<<<grid, 128, 0, stream>>>(he, w2_bf, b2, nullptr, xe,
        D_MODEL, D_INNER,
        (long)CAP_PAD * D_INNER, (long)D_INNER * D_MODEL, N_EXP,
        (long)D_MODEL, (long)CAP_PAD * D_MODEL, 0);
  }
  combine_kernel<<<(unsigned)NTOK, 256, 0, stream>>>(xe, e1, e2, slot1, slot2, wn1, wn2,
                                                     y_g, D_MODEL);

  // 9) ungroup + residual + final LN -> d_out
  final_ln_kernel<<<(unsigned)ROWS, 256, 0, stream>>>(y_g, ctx_ln, ln_out_g, ln_out_b, out);
}
